// Decoder_1108101563082
// MI455X (gfx1250) — compile-verified, matrix operand order checked by reading, not run on hardware
//
#include <hip/hip_runtime.h>
#include <hip/hip_bf16.h>
#include <stdint.h>

// Problem constants (from reference)
#define NN   21
#define BB   256
#define PHN  12
#define K0P  352    // IN0=328 padded to multiple of 32
#define KH   512
#define GXH  1536
#define GH   512
#define GO   128
#define NTY  4

typedef __attribute__((ext_vector_type(16))) __bf16 v16bf;
typedef __attribute__((ext_vector_type(8)))  float  v8f;

union Frag { uint4 u[2]; v16bf v; };

__device__ __forceinline__ unsigned short f2bf(float f) {
  unsigned u = __float_as_uint(f);
  u += 0x7FFFu + ((u >> 16) & 1u);   // round-to-nearest-even
  return (unsigned short)(u >> 16);
}
__device__ __forceinline__ float sigmoidf_(float x) { return 1.f / (1.f + __expf(-x)); }

// ---- Convert fp32 weights [t][g][Kin] -> bf16 [t][g][Kpad] (zero pad) ----
__global__ void conv_w(const float* __restrict__ src, unsigned short* __restrict__ dst,
                       int G, int Kin, int Kpad, int total) {
  int i = blockIdx.x * blockDim.x + threadIdx.x;
  if (i >= total) return;
  int k = i % Kpad;
  int g = (i / Kpad) % G;
  int t = i / (Kpad * G);
  dst[i] = (k < Kin) ? f2bf(src[((size_t)t * G + g) * Kin + k]) : (unsigned short)0;
}

// ---- L1-normalize graph rows ----
__global__ void norm_graphs(const float* __restrict__ G0, const float* __restrict__ Grnn,
                            const float* __restrict__ GA, const float* __restrict__ Gq,
                            const float* __restrict__ Gc,
                            float* __restrict__ G0n, float* __restrict__ Gnn,
                            float* __restrict__ Gqn, float* __restrict__ Gcn) {
  int n = threadIdx.x;
  if (n >= NN) return;
  float s0 = 0.f, sr = 0.f, sq = 0.f, sc = 0.f;
  for (int m = 0; m < NN; m++) {
    s0 += fabsf(G0[n * NN + m]);  sr += fabsf(Grnn[n * NN + m]);
    sq += fabsf(Gq[n * NN + m]);  sc += fabsf(Gc[n * NN + m]);
  }
  s0 = fmaxf(s0, 1e-12f); sr = fmaxf(sr, 1e-12f);
  sq = fmaxf(sq, 1e-12f); sc = fmaxf(sc, 1e-12f);
  for (int m = 0; m < NN; m++) {
    G0n[n * NN + m] = G0[n * NN + m] / s0;
    Gnn[n * NN + m] = Grnn[n * NN + m] / sr + GA[n * NN + m];
    Gqn[n * NN + m] = Gq[n * NN + m] / sq;
    Gcn[n * NN + m] = Gc[n * NN + m] / sc;
  }
}

// ---- Build bf16 activation panels: xin (x_t | z | h) and a0 (x_{t-1} | z | h) ----
__global__ void fill_A(const float* __restrict__ x, const float* __restrict__ h,
                       const float* __restrict__ z,
                       unsigned short* __restrict__ xin, unsigned short* __restrict__ a0) {
  int i = blockIdx.x * blockDim.x + threadIdx.x;
  if (i >= NN * BB * K0P) return;
  int k = i % K0P;
  int b = (i / K0P) % BB;
  int n = i / (K0P * BB);
  unsigned short vt = 0, vt1 = 0;
  if (k < 8) {
    vt  = f2bf(x[(((size_t)b * 2 + 1) * NN + n) * 8 + k]);
    vt1 = f2bf(x[(((size_t)b * 2 + 0) * NN + n) * 8 + k]);
  } else if (k < 72) {
    unsigned short v = f2bf(z[((size_t)b * NN + n) * 64 + (k - 8)]);
    vt = v; vt1 = v;
  } else if (k < 328) {
    unsigned short v = f2bf(h[((size_t)b * NN + n) * 256 + (k - 72)]);
    vt = v; vt1 = v;
  }
  xin[i] = vt; a0[i] = vt1;
}

// ---- init q_c state and x_t output section ----
__global__ void init_misc(const float* __restrict__ x, const float* __restrict__ q_t,
                          float* __restrict__ out, float* __restrict__ qc) {
  int i = blockIdx.x * blockDim.x + threadIdx.x;
  if (i >= BB * NN * 8) return;
  int f = i % 8;
  int n = (i / 8) % NN;
  int b = i / (8 * NN);
  out[(size_t)BB * PHN * NN * 10 + ((size_t)b * NN + n) * 8 + f] =
      x[(((size_t)b * 2 + 1) * NN + n) * 8 + f];
  if (f < 4) qc[((size_t)b * NN + n) * 4 + f] = q_t[((size_t)b * NN + n) * 4 + f];
}

// ---- Per-node batched GEMM via WMMA bf16->f32 ----
// A: [NN][BB][Kpad] bf16, W: [NTY][Gout][Kpad] bf16 (row g = column of B),
// P[node][m][g] = sum_k A*W + bias[node][g]
// Block: 4 waves as 2x2 grid, block tile 64(M) x 128(N); wave tile 32x64
// (2 A-frags x 4 B-frags, 8 f32 accumulators -> ~22 FLOP/B from L2).
__global__ __launch_bounds__(128)
void gemm_node(const unsigned short* __restrict__ A, const unsigned short* __restrict__ W,
               const float* __restrict__ bias, const int* __restrict__ T,
               float* __restrict__ P, int Kpad, int Gout) {
  const int node = blockIdx.z;
  const int wave = threadIdx.x >> 5;
  const int wm   = wave >> 1;       // 0..1 : M sub-tile
  const int wn   = wave & 1;        // 0..1 : N sub-tile
  const int lane = threadIdx.x & 31;
  const int half = lane >> 4;       // K-half select (ISA 16-bit A layout)
  const int row  = lane & 15;       // M row (A) / N col (B)
  const int m0 = blockIdx.y * 64 + wm * 32;
  const int g0 = blockIdx.x * 128 + wn * 64;
  const int t  = T[node];

  const unsigned short* Abase = A + ((size_t)node * BB + m0 + row) * Kpad + half * 8;
  const unsigned short* Wbase = W + ((size_t)t * Gout + g0 + row) * Kpad + half * 8;

  v8f acc[2][4] = {};
  for (int k = 0; k < Kpad; k += 32) {
    Frag a[2], b[4];
#pragma unroll
    for (int i = 0; i < 2; ++i) {
      const unsigned short* p = Abase + k + (size_t)(16 * i) * Kpad;
      a[i].u[0] = *(const uint4*)p;
      a[i].u[1] = *(const uint4*)(p + 16);
    }
#pragma unroll
    for (int j = 0; j < 4; ++j) {
      const unsigned short* p = Wbase + k + (size_t)(16 * j) * Kpad;
      b[j].u[0] = *(const uint4*)p;
      b[j].u[1] = *(const uint4*)(p + 16);
    }
#pragma unroll
    for (int i = 0; i < 2; ++i)
#pragma unroll
      for (int j = 0; j < 4; ++j)
        acc[i][j] = __builtin_amdgcn_wmma_f32_16x16x32_bf16(
            false, a[i].v, false, b[j].v, (short)0, acc[i][j], false, false);
  }

  const float* bb = bias + (size_t)node * Gout;
  float* Pb = P + (size_t)node * BB * Gout;
#pragma unroll
  for (int j = 0; j < 4; ++j) {
    const int col = g0 + 16 * j + row;
    const float bv = bb[col];
#pragma unroll
    for (int i = 0; i < 2; ++i) {
#pragma unroll
      for (int v = 0; v < 8; ++v) {
        int m = m0 + 16 * i + half * 8 + v;   // C/D layout: lanes16-31 -> M+8
        Pb[(size_t)m * Gout + col] = acc[i][j][v] + bv;
      }
    }
  }
}

// ---- h0 = G0n-mix of P0 ----
__global__ void rnn_mix(const float* __restrict__ G0n, const float* __restrict__ P0,
                        float* __restrict__ hf, unsigned short* __restrict__ hb) {
  int i = blockIdx.x * blockDim.x + threadIdx.x;
  if (i >= NN * BB * GH) return;
  int g = i % GH;
  int b = (i / GH) % BB;
  int n = i / (GH * BB);
  float s = 0.f;
  size_t base = (size_t)b * GH + g;
  for (int m = 0; m < NN; m++) s += G0n[n * NN + m] * P0[(size_t)m * BB * GH + base];
  hf[i] = s; hb[i] = f2bf(s);
}

// ---- Graph mix of Px/Ph + GRU gate math ----
__global__ void gru_mix(const float* __restrict__ Gn, const float* __restrict__ Px,
                        const float* __restrict__ Ph, float* __restrict__ hf,
                        unsigned short* __restrict__ hb, unsigned short* __restrict__ yb) {
  int i = blockIdx.x * blockDim.x + threadIdx.x;
  if (i >= NN * BB * GH) return;
  int g = i % GH;
  int b = (i / GH) % BB;
  int n = i / (GH * BB);
  float xr = 0.f, xz = 0.f, xc = 0.f, hr = 0.f, hz = 0.f, hc2 = 0.f;
  size_t base = (size_t)b * GXH + g;
  for (int m = 0; m < NN; m++) {
    float w = Gn[n * NN + m];
    size_t p = (size_t)m * BB * GXH + base;
    xr += w * Px[p]; xz += w * Px[p + 512]; xc += w * Px[p + 1024];
    hr += w * Ph[p]; hz += w * Ph[p + 512]; hc2 += w * Ph[p + 1024];
  }
  float r  = sigmoidf_(xr + hr);
  float zg = sigmoidf_(xz + hz);
  float nn_ = tanhf(xc + r * hc2);
  float hc = hf[i];
  float hn = (1.f - zg) * nn_ + zg * hc;
  hf[i] = hn;
  hb[i] = f2bf(hn);
  yb[i] = f2bf(tanhf(hn));
}

// ---- ys/yc = tanh(graph mix of head GEMM results) ----
__global__ void ysyc_mix(const float* __restrict__ Gqn, const float* __restrict__ Gcn,
                         const float* __restrict__ Qp, const float* __restrict__ Cp,
                         float* __restrict__ ys, float* __restrict__ yc) {
  int i = blockIdx.x * blockDim.x + threadIdx.x;
  if (i >= NN * BB * GO) return;
  int o = i % GO;
  int b = (i / GO) % BB;
  int n = i / (GO * BB);
  float sq = 0.f, sc = 0.f;
  size_t base = (size_t)b * GO + o;
  for (int m = 0; m < NN; m++) {
    size_t p = (size_t)m * BB * GO + base;
    sq += Gqn[n * NN + m] * Qp[p];
    sc += Gcn[n * NN + m] * Cp[p];
  }
  ys[i] = tanhf(sq);
  yc[i] = tanhf(sc);
}

// ---- quaternion head: dq3/cov dots, quat_exp, qmul, outputs, next-step x_c ----
__global__ void head_k(const float* __restrict__ ys, const float* __restrict__ yc,
                       const float* __restrict__ W3, const float* __restrict__ b3,
                       const float* __restrict__ W6, const float* __restrict__ b6,
                       const int* __restrict__ T, float* __restrict__ qc,
                       unsigned short* __restrict__ xin, float* __restrict__ out, int step) {
  int i = blockIdx.x * blockDim.x + threadIdx.x;
  if (i >= BB * NN) return;
  int n = i % NN;
  int b = i / NN;
  int t = T[n];
  const float* ysp = ys + ((size_t)n * BB + b) * GO;
  const float* ycp = yc + ((size_t)n * BB + b) * GO;
  float d0 = b3[n * 3 + 0], d1 = b3[n * 3 + 1], d2 = b3[n * 3 + 2];
  float c0 = b6[n * 6 + 0], c1 = b6[n * 6 + 1], c2 = b6[n * 6 + 2];
  float c3 = b6[n * 6 + 3], c4 = b6[n * 6 + 4], c5 = b6[n * 6 + 5];
  const float* w3 = W3 + (size_t)t * 3 * GO;
  const float* w6 = W6 + (size_t)t * 6 * GO;
  for (int o = 0; o < GO; o++) {
    float a = ysp[o], cv = ycp[o];
    d0 += a * w3[o];          d1 += a * w3[GO + o];     d2 += a * w3[2 * GO + o];
    c0 += cv * w6[o];         c1 += cv * w6[GO + o];    c2 += cv * w6[2 * GO + o];
    c3 += cv * w6[3 * GO + o]; c4 += cv * w6[4 * GO + o]; c5 += cv * w6[5 * GO + o];
  }
  float theta = sqrtf(d0 * d0 + d1 * d1 + d2 * d2);
  float w = cosf(0.5f * theta);
  float kk = (theta > 1e-8f) ? (sinf(0.5f * theta) / fmaxf(theta, 1e-8f)) : 0.5f;
  float dqw = w, dqx = kk * d0, dqy = kk * d1, dqz = kk * d2;
  float* q = qc + ((size_t)b * NN + n) * 4;
  float qw = q[0], qx = q[1], qy = q[2], qz = q[3];
  float nw = dqw * qw - (dqx * qx + dqy * qy + dqz * qz);
  float nx = dqw * qx + qw * dqx + (dqy * qz - dqz * qy);
  float ny = dqw * qy + qw * dqy + (dqz * qx - dqx * qz);
  float nz = dqw * qz + qw * dqz + (dqx * qy - dqy * qx);
  q[0] = nw; q[1] = nx; q[2] = ny; q[3] = nz;
  size_t odq = (((size_t)b * PHN + step) * NN + n) * 4;
  out[odq + 0] = dqw; out[odq + 1] = dqx; out[odq + 2] = dqy; out[odq + 3] = dqz;
  size_t ocv = (size_t)BB * PHN * NN * 4 + (((size_t)b * PHN + step) * NN + n) * 6;
  out[ocv + 0] = c0; out[ocv + 1] = c1; out[ocv + 2] = c2;
  out[ocv + 3] = c3; out[ocv + 4] = c4; out[ocv + 5] = c5;
  unsigned short* xp = xin + ((size_t)n * BB + b) * K0P;
  xp[0] = f2bf(nw);  xp[1] = f2bf(nx);  xp[2] = f2bf(ny);  xp[3] = f2bf(nz);
  xp[4] = f2bf(dqw); xp[5] = f2bf(dqx); xp[6] = f2bf(dqy); xp[7] = f2bf(dqz);
}

extern "C" void kernel_launch(void* const* d_in, const int* in_sizes, int n_in,
                              void* d_out, int out_size, void* d_ws, size_t ws_size,
                              hipStream_t stream) {
  (void)in_sizes; (void)n_in; (void)out_size; (void)ws_size;
  const float* x    = (const float*)d_in[0];
  const float* h    = (const float*)d_in[1];
  const float* z    = (const float*)d_in[2];
  const float* q_t  = (const float*)d_in[3];
  const int*   T    = (const int*)d_in[4];
  const float* W0   = (const float*)d_in[6];
  const float* b0   = (const float*)d_in[7];
  const float* G0   = (const float*)d_in[8];
  const float* Wih  = (const float*)d_in[9];
  const float* Whh  = (const float*)d_in[10];
  const float* bih  = (const float*)d_in[11];
  const float* bhh  = (const float*)d_in[12];
  const float* Grnn = (const float*)d_in[13];
  const float* GA   = (const float*)d_in[14];
  const float* Wq   = (const float*)d_in[15];
  const float* bq   = (const float*)d_in[16];
  const float* Gq   = (const float*)d_in[17];
  const float* Wc   = (const float*)d_in[18];
  const float* bc   = (const float*)d_in[19];
  const float* Gc   = (const float*)d_in[20];
  const float* W3   = (const float*)d_in[21];
  const float* b3   = (const float*)d_in[22];
  const float* W6   = (const float*)d_in[23];
  const float* b6   = (const float*)d_in[24];
  float* out = (float*)d_out;

  uint8_t* ws = (uint8_t*)d_ws;
  size_t off = 0;
  auto alloc = [&](size_t bytes) -> void* {
    size_t o = (off + 255) & ~(size_t)255;
    off = o + bytes;
    return (void*)(ws + o);
  };
  unsigned short* Wih_b = (unsigned short*)alloc((size_t)NTY * GXH * K0P * 2);
  unsigned short* Whh_b = (unsigned short*)alloc((size_t)NTY * GXH * KH * 2);
  unsigned short* W0_b  = (unsigned short*)alloc((size_t)NTY * GH * K0P * 2);
  unsigned short* Wq_b  = (unsigned short*)alloc((size_t)NTY * GO * KH * 2);
  unsigned short* Wc_b  = (unsigned short*)alloc((size_t)NTY * GO * KH * 2);
  unsigned short* xin_b = (unsigned short*)alloc((size_t)NN * BB * K0P * 2);
  unsigned short* a0_b  = (unsigned short*)alloc((size_t)NN * BB * K0P * 2);
  unsigned short* h_b   = (unsigned short*)alloc((size_t)NN * BB * KH * 2);
  unsigned short* y_b   = (unsigned short*)alloc((size_t)NN * BB * KH * 2);
  float* h_f  = (float*)alloc((size_t)NN * BB * KH * 4);
  float* Px   = (float*)alloc((size_t)NN * BB * GXH * 4);
  float* Ph   = (float*)alloc((size_t)NN * BB * GXH * 4);
  float* Qp   = (float*)alloc((size_t)NN * BB * GO * 4);
  float* Cp   = (float*)alloc((size_t)NN * BB * GO * 4);
  float* ysb  = (float*)alloc((size_t)NN * BB * GO * 4);
  float* ycb  = (float*)alloc((size_t)NN * BB * GO * 4);
  float* qc   = (float*)alloc((size_t)BB * NN * 4 * 4);
  float* G0n  = (float*)alloc((size_t)NN * NN * 4);
  float* Gnn  = (float*)alloc((size_t)NN * NN * 4);
  float* Gqn  = (float*)alloc((size_t)NN * NN * 4);
  float* Gcn  = (float*)alloc((size_t)NN * NN * 4);

  // --- one-time conversions / setup ---
  {
    int tot;
    tot = NTY * GXH * K0P;
    conv_w<<<(tot + 255) / 256, 256, 0, stream>>>(Wih, Wih_b, GXH, 328, K0P, tot);
    tot = NTY * GXH * KH;
    conv_w<<<(tot + 255) / 256, 256, 0, stream>>>(Whh, Whh_b, GXH, 512, KH, tot);
    tot = NTY * GH * K0P;
    conv_w<<<(tot + 255) / 256, 256, 0, stream>>>(W0, W0_b, GH, 328, K0P, tot);
    tot = NTY * GO * KH;
    conv_w<<<(tot + 255) / 256, 256, 0, stream>>>(Wq, Wq_b, GO, 512, KH, tot);
    conv_w<<<(tot + 255) / 256, 256, 0, stream>>>(Wc, Wc_b, GO, 512, KH, tot);
  }
  norm_graphs<<<1, 32, 0, stream>>>(G0, Grnn, GA, Gq, Gc, G0n, Gnn, Gqn, Gcn);
  {
    int tot = NN * BB * K0P;
    fill_A<<<(tot + 255) / 256, 256, 0, stream>>>(x, h, z, xin_b, a0_b);
  }
  init_misc<<<(BB * NN * 8 + 255) / 256, 256, 0, stream>>>(x, q_t, out, qc);

  // --- initial hidden state: rnn_h = G0-mix(A_init @ W0 + b0) ---
  {
    dim3 g(GH / 128, BB / 64, NN);
    gemm_node<<<g, 128, 0, stream>>>(a0_b, W0_b, b0, T, Px, K0P, GH);
    int tot = NN * BB * GH;
    rnn_mix<<<(tot + 255) / 256, 256, 0, stream>>>(G0n, Px, h_f, h_b);
  }

  // --- rollout ---
  for (int step = 0; step < PHN; ++step) {
    dim3 gx(GXH / 128, BB / 64, NN);
    gemm_node<<<gx, 128, 0, stream>>>(xin_b, Wih_b, bih, T, Px, K0P, GXH);
    gemm_node<<<gx, 128, 0, stream>>>(h_b,   Whh_b, bhh, T, Ph, KH,  GXH);
    {
      int tot = NN * BB * GH;
      gru_mix<<<(tot + 255) / 256, 256, 0, stream>>>(Gnn, Px, Ph, h_f, h_b, y_b);
    }
    dim3 go(GO / 128, BB / 64, NN);
    gemm_node<<<go, 128, 0, stream>>>(y_b, Wq_b, bq, T, Qp, KH, GO);
    gemm_node<<<go, 128, 0, stream>>>(y_b, Wc_b, bc, T, Cp, KH, GO);
    {
      int tot = NN * BB * GO;
      ysyc_mix<<<(tot + 255) / 256, 256, 0, stream>>>(Gqn, Gcn, Qp, Cp, ysb, ycb);
    }
    head_k<<<(BB * NN + 127) / 128, 128, 0, stream>>>(ysb, ycb, W3, b3, W6, b6, T, qc,
                                                      xin_b, out, step);
  }
}
